// NormalizedMamba2Layer_38706245271932
// MI455X (gfx1250) — matrix operand order repaired and checked by reading, not compiled, not run on hardware
//
#include <hip/hip_runtime.h>
#include <hip/hip_bf16.h>

typedef __attribute__((ext_vector_type(16))) _Float16 v16h;
typedef __attribute__((ext_vector_type(8)))  _Float16 v8h;
typedef __attribute__((ext_vector_type(8)))  float    v8f;

#define DIM       1024
#define D_STATE   64
#define D_CONV    4
#define HEADDIM   128
#define D_INNER   2048
#define NHEADS    16
#define CONV_DIM  2176
#define D_IN_PROJ 4240
#define NPAD_IN   4352          // D_IN_PROJ padded to multiple of 64 for tiling
#define CHUNK     256
#define BATCH     2
#define SEQLEN    4096
#define NCHUNK    (SEQLEN / CHUNK)   // 16
#define NBC       (BATCH * NCHUNK)   // 32
#define MTOT      (BATCH * SEQLEN)   // 8192
#define LN_EPS    1e-5f
#define RMS_EPS   1e-5f

// ---------------------------------------------------------------------------
// Small device helpers
// ---------------------------------------------------------------------------
__device__ __forceinline__ float sigmoidf_(float x) { return 1.0f / (1.0f + __expf(-x)); }
__device__ __forceinline__ float siluf_(float x)    { return x * sigmoidf_(x); }
__device__ __forceinline__ float softplusf_(float x){ return (x > 20.0f) ? x : log1pf(__expf(x)); }

// CDNA5 async global->LDS copy (ASYNCcnt-tracked), 16B per lane.
// VDST = LDS byte address (low 32 bits of the generic pointer), VADDR = global.
__device__ __forceinline__ void async_copy_b128(const void* gsrc, void* ldst)
{
    asm volatile("global_load_async_to_lds_b128 %0, %1, off"
                 :: "v"((unsigned)(unsigned long long)ldst), "v"(gsrc)
                 : "memory");
}
__device__ __forceinline__ void wait_async_le3()
{
    asm volatile("s_wait_asynccnt 0x3" ::: "memory");
}
__device__ __forceinline__ void wait_async_le0()
{
    asm volatile("s_wait_asynccnt 0x0" ::: "memory");
}

// ---------------------------------------------------------------------------
// 1) Row-wise L2 normalize + optional scale, cast f32 -> f16. Rows >= O zeroed
//    (padding rows so the GEMM can run on a padded N without bounds checks).
// ---------------------------------------------------------------------------
__global__ __launch_bounds__(256) void l2norm_cast_kernel(
    const float* __restrict__ src, _Float16* __restrict__ dst,
    int O, int K, float scale)
{
    __shared__ float red[256];
    const int row = blockIdx.x;
    const int tid = threadIdx.x;
    if (row >= O) {
        for (int c = tid; c < K; c += 256) dst[(long)row * K + c] = (_Float16)0.0f;
        return;
    }
    const float* sr = src + (long)row * K;
    float s = 0.0f;
    for (int c = tid; c < K; c += 256) { float w = sr[c]; s += w * w; }
    red[tid] = s; __syncthreads();
    for (int o = 128; o > 0; o >>= 1) { if (tid < o) red[tid] += red[tid + o]; __syncthreads(); }
    const float inv = scale / fmaxf(sqrtf(red[0]), 1e-6f);
    for (int c = tid; c < K; c += 256) dst[(long)row * K + c] = (_Float16)(sr[c] * inv);
}

// ---------------------------------------------------------------------------
// 2) LayerNorm over DIM, cast to f16
// ---------------------------------------------------------------------------
__global__ __launch_bounds__(256) void layernorm_cast_kernel(
    const float* __restrict__ u, const float* __restrict__ w,
    const float* __restrict__ b, _Float16* __restrict__ dst)
{
    __shared__ float red[256];
    const int row = blockIdx.x;
    const int tid = threadIdx.x;
    const float* ur = u + (long)row * DIM;
    float loc[DIM / 256];
    float s = 0.0f;
#pragma unroll
    for (int k = 0; k < DIM / 256; ++k) { loc[k] = ur[tid + k * 256]; s += loc[k]; }
    red[tid] = s; __syncthreads();
    for (int o = 128; o > 0; o >>= 1) { if (tid < o) red[tid] += red[tid + o]; __syncthreads(); }
    const float mu = red[0] * (1.0f / DIM);
    __syncthreads();
    float v = 0.0f;
#pragma unroll
    for (int k = 0; k < DIM / 256; ++k) { float d = loc[k] - mu; v += d * d; }
    red[tid] = v; __syncthreads();
    for (int o = 128; o > 0; o >>= 1) { if (tid < o) red[tid] += red[tid + o]; __syncthreads(); }
    const float rs = rsqrtf(red[0] * (1.0f / DIM) + LN_EPS);
#pragma unroll
    for (int k = 0; k < DIM / 256; ++k) {
        int c = tid + k * 256;
        dst[(long)row * DIM + c] = (_Float16)((loc[k] - mu) * rs * w[c] + b[c]);
    }
}

// ---------------------------------------------------------------------------
// 3) Generic WMMA f16 GEMM:  C[m,n] = sum_k A[m,k] * Bt[n,k]   (both row-major
//    over K). Block = 256 threads = 8 waves; tile 128(M) x 64(N) x 32(K).
//    Wave grid 4(M) x 2(N), each wave 32x32 via 2x2 v_wmma_f32_16x16x32_f16.
//    Double-buffered LDS staging via GLOBAL_LOAD_ASYNC_TO_LDS_B128: each
//    thread issues exactly 3 async b128 ops per K tile (uniform ASYNCcnt +3
//    per wave), so `s_wait_asynccnt 3` releases the previous tile while the
//    next tile's DMA is still in flight. Batched via blockIdx.z.
// ---------------------------------------------------------------------------
__global__ __launch_bounds__(256) void wmma_gemm_kernel(
    const _Float16* __restrict__ A, const _Float16* __restrict__ Bt,
    float* __restrict__ C, int K, int ldc,
    long batchA, long batchB, long batchC)
{
    __shared__ _Float16 As[2][128][40];
    __shared__ _Float16 Bs[2][64][40];
    A  += (long)blockIdx.z * batchA;
    Bt += (long)blockIdx.z * batchB;
    C  += (long)blockIdx.z * batchC;
    const int tid  = threadIdx.x;
    const int lane = tid & 31;
    const int wave = tid >> 5;
    const int wm = wave >> 1;              // 0..3
    const int wn = wave & 1;               // 0..1
    const int bm = blockIdx.x * 128;
    const int bn = blockIdx.y * 64;

    v8f zero = {};
    v8f acc[2][2];
    acc[0][0] = zero; acc[0][1] = zero; acc[1][0] = zero; acc[1][1] = zero;

    const int arow  = tid >> 1;            // 0..127, 2 x 16B per thread
    const int aseg  = (tid & 1) * 16;
    const int brow4 = tid >> 2;            // 0..63,  1 x 16B per thread
    const int bseg4 = (tid & 3) * 8;
    const int kb8  = (lane >> 4) * 8;
    const int kb16 = (lane >> 4) * 16;
    const int l15  = lane & 15;

    const _Float16* arowp = A  + (long)(bm + arow) * K + aseg;
    const _Float16* browp = Bt + (long)(bn + brow4) * K + bseg4;

    // prologue: stage tile 0 into buffer 0
    async_copy_b128(arowp,     &As[0][arow][aseg]);
    async_copy_b128(arowp + 8, &As[0][arow][aseg + 8]);
    async_copy_b128(browp,     &Bs[0][brow4][bseg4]);

    int cur = 0;
    for (int kt = 0; kt < K; kt += 32) {
        const bool more = (kt + 32) < K;
        if (more) {   // stage next tile into the other buffer
            const _Float16* an = arowp + kt + 32;
            const _Float16* bn2 = browp + kt + 32;
            async_copy_b128(an,     &As[cur ^ 1][arow][aseg]);
            async_copy_b128(an + 8, &As[cur ^ 1][arow][aseg + 8]);
            async_copy_b128(bn2,    &Bs[cur ^ 1][brow4][bseg4]);
            wait_async_le3();          // current tile's 3 ops complete
        } else {
            wait_async_le0();
        }
        __syncthreads();
#pragma unroll
        for (int mt = 0; mt < 2; ++mt) {
            const int row = wm * 32 + mt * 16 + l15;
            v16h a;
#pragma unroll
            for (int i = 0; i < 16; ++i) a[i] = As[cur][row][kb16 + i];
#pragma unroll
            for (int nt = 0; nt < 2; ++nt) {
                const int col = wn * 32 + nt * 16 + l15;
                v16h b;
#pragma unroll
                for (int i = 0; i < 16; ++i) b[i] = Bs[cur][col][kb16 + i];
                acc[mt][nt] = __builtin_amdgcn_wmma_f32_16x16x32_f16(
                    false, a, false, b, (short)0, acc[mt][nt], false, false);
            }
        }
        __syncthreads();   // all reads of buf `cur` done before it is re-staged
        cur ^= 1;
    }
#pragma unroll
    for (int mt = 0; mt < 2; ++mt)
#pragma unroll
    for (int nt = 0; nt < 2; ++nt) {
        const int col   = bn + wn * 32 + nt * 16 + l15;
        const int rbase = bm + wm * 32 + mt * 16 + (lane >> 4) * 8;
#pragma unroll
        for (int r = 0; r < 8; ++r)
            C[(long)(rbase + r) * ldc + col] = acc[mt][nt][r];
    }
}

// ---------------------------------------------------------------------------
// 4) Depthwise causal conv (width 4) + bias + SiLU; split & cast to f16
// ---------------------------------------------------------------------------
__global__ __launch_bounds__(256) void conv_silu_kernel(
    const float* __restrict__ zx, const float* __restrict__ convw,
    const float* __restrict__ convb,
    _Float16* __restrict__ xf, _Float16* __restrict__ Bf, _Float16* __restrict__ Cf)
{
    const long idx = (long)blockIdx.x * 256 + threadIdx.x;
    if (idx >= (long)MTOT * CONV_DIM) return;
    const int  ch  = (int)(idx % CONV_DIM);
    const long row = idx / CONV_DIM;
    const int  b   = (int)(row / SEQLEN);
    const int  t   = (int)(row % SEQLEN);
    float s = convb[ch];
#pragma unroll
    for (int j = 0; j < D_CONV; ++j) {
        int tt = t - (D_CONV - 1) + j;
        if (tt >= 0)
            s += convw[ch * D_CONV + j] * zx[((long)b * SEQLEN + tt) * NPAD_IN + D_INNER + ch];
    }
    s = siluf_(s);
    if (ch < D_INNER)               xf[row * D_INNER + ch] = (_Float16)s;
    else if (ch < D_INNER + D_STATE) Bf[row * D_STATE + (ch - D_INNER)] = (_Float16)s;
    else                             Cf[row * D_STATE + (ch - D_INNER - D_STATE)] = (_Float16)s;
}

// ---------------------------------------------------------------------------
// 5) dt = softplus(dt_raw + dt_bias)
// ---------------------------------------------------------------------------
__global__ __launch_bounds__(256) void dt_softplus_kernel(
    const float* __restrict__ zx, const float* __restrict__ dt_bias,
    float* __restrict__ dt)
{
    const long idx = (long)blockIdx.x * 256 + threadIdx.x;
    if (idx >= (long)MTOT * NHEADS) return;
    const int  h   = (int)(idx % NHEADS);
    const long row = idx / NHEADS;
    const float v  = zx[row * NPAD_IN + (D_INNER + CONV_DIM) + h] + dt_bias[h];
    dt[row * NHEADS + h] = softplusf_(v);
}

// ---------------------------------------------------------------------------
// 6) Per (chunk, head) inclusive cumsum of dA = dt * A  (Hillis-Steele in LDS)
//    cum layout: [NBC][NHEADS][CHUNK]
// ---------------------------------------------------------------------------
__global__ __launch_bounds__(256) void cumsum_kernel(
    const float* __restrict__ dt, const float* __restrict__ A_log,
    float* __restrict__ cum)
{
    __shared__ float s[CHUNK];
    const int bc = blockIdx.x, h = blockIdx.y, t = threadIdx.x;
    const float A = -__expf(A_log[h]);
    s[t] = dt[((long)bc * CHUNK + t) * NHEADS + h] * A;
    __syncthreads();
    for (int off = 1; off < CHUNK; off <<= 1) {
        float v = (t >= off) ? s[t - off] : 0.0f;
        __syncthreads();
        s[t] += v;
        __syncthreads();
    }
    cum[((long)bc * NHEADS + h) * CHUNK + t] = s[t];
}

// ---------------------------------------------------------------------------
// 7) Causal intra-chunk  y = att @ x  per (half-chunk, chunk, head).
//    att[i][j] = scores[i][j] * exp(cum[i]-cum[j]) * dt[j]  (j <= i),
//    materialized in 32-wide K slabs into LDS as f16, then WMMA against an
//    LDS-transposed x slab. Block tile 128(i) x 128(p); wave 32 x 64.
// ---------------------------------------------------------------------------
__global__ __launch_bounds__(256) void y_intra_kernel(
    const float*   __restrict__ scores,  // [NBC][CHUNK][CHUNK]
    const float*   __restrict__ cum,     // [NBC][NHEADS][CHUNK]
    const float*   __restrict__ dt,      // [MTOT][NHEADS]
    const _Float16* __restrict__ xf,     // [MTOT][D_INNER]
    float* __restrict__ y)               // [MTOT][D_INNER]
{
    __shared__ _Float16 attS[128][40];   // [i][j]
    __shared__ _Float16 xS[128][40];     // [p][j]
    const int mhalf = blockIdx.x;        // 0..1
    const int bc    = blockIdx.y;        // 0..31
    const int h     = blockIdx.z;        // 0..15
    const int tid = threadIdx.x, lane = tid & 31, wave = tid >> 5;
    const int wm = wave >> 1, wn = wave & 1;
    const int mbase = mhalf * 128;
    const long cb = (long)bc * CHUNK;
    const float* cumh = cum + ((long)bc * NHEADS + h) * CHUNK;
    const int kb8 = (lane >> 4) * 8, kb16 = (lane >> 4) * 16, l15 = lane & 15;

    v8f zero = {};
    v8f acc[2][4];
#pragma unroll
    for (int i = 0; i < 2; ++i)
#pragma unroll
    for (int j = 0; j < 4; ++j) acc[i][j] = zero;

    for (int kb = 0; kb < mbase + 128; kb += 32) {
        // prefetch next scores slab into L2/L0 while we build this one
        if (kb + 32 < mbase + 128)
            __builtin_prefetch(&scores[((long)bc * CHUNK + mbase + (tid >> 5)) * CHUNK + kb + 32], 0, 1);
#pragma unroll
        for (int q = 0; q < 16; ++q) {        // att slab 128x32
            const int idx = tid + q * 256;
            const int i = idx >> 5, j = idx & 31;
            const int gi = mbase + i, gj = kb + j;
            float v = 0.0f;
            if (gj <= gi) {
                const float sc = scores[((long)bc * CHUNK + gi) * CHUNK + gj];
                v = sc * __expf(cumh[gi] - cumh[gj]) * dt[(cb + gj) * NHEADS + h];
            }
            attS[i][j] = (_Float16)v;
        }
#pragma unroll
        for (int q = 0; q < 16; ++q) {        // x slab transposed: [p][j]
            const int idx = tid + q * 256;
            const int j = idx >> 7, p = idx & 127;
            xS[p][j] = xf[(cb + kb + j) * (long)D_INNER + h * HEADDIM + p];
        }
        __syncthreads();
#pragma unroll
        for (int mt = 0; mt < 2; ++mt) {
            const int row = wm * 32 + mt * 16 + l15;
            v16h a;
#pragma unroll
            for (int i = 0; i < 16; ++i) a[i] = attS[row][kb16 + i];
#pragma unroll
            for (int nt = 0; nt < 4; ++nt) {
                const int col = wn * 64 + nt * 16 + l15;
                v16h b;
#pragma unroll
                for (int i = 0; i < 16; ++i) b[i] = xS[col][kb16 + i];
                acc[mt][nt] = __builtin_amdgcn_wmma_f32_16x16x32_f16(
                    false, a, false, b, (short)0, acc[mt][nt], false, false);
            }
        }
        __syncthreads();
    }
#pragma unroll
    for (int mt = 0; mt < 2; ++mt)
#pragma unroll
    for (int nt = 0; nt < 4; ++nt) {
        const int gp    = wn * 64 + nt * 16 + l15;
        const int rbase = mbase + wm * 32 + mt * 16 + (lane >> 4) * 8;
#pragma unroll
        for (int r = 0; r < 8; ++r)
            y[(cb + rbase + r) * (long)D_INNER + h * HEADDIM + gp] = acc[mt][nt][r];
    }
}

// ---------------------------------------------------------------------------
// 8) Chunk states: states[p,n] = sum_l (x[l,p] * dt[l] * exp(cumEnd-cum[l])) * B[l,n]
//    M=128(p), N=64(n), K=256(l). Wave 32x32, 2x2 accumulators.
// ---------------------------------------------------------------------------
__global__ __launch_bounds__(256) void states_kernel(
    const _Float16* __restrict__ Bf,   // [MTOT][D_STATE]
    const _Float16* __restrict__ xf,   // [MTOT][D_INNER]
    const float* __restrict__ cum, const float* __restrict__ dt,
    float* __restrict__ states)        // [NBC][NHEADS][HEADDIM][D_STATE]
{
    __shared__ _Float16 aS[128][40];   // [p][j]
    __shared__ _Float16 bS[64][40];    // [n][j]
    const int bc = blockIdx.x, h = blockIdx.y;
    const int tid = threadIdx.x, lane = tid & 31, wave = tid >> 5;
    const int wm = wave >> 1, wn = wave & 1;
    const long cb = (long)bc * CHUNK;
    const float* cumh = cum + ((long)bc * NHEADS + h) * CHUNK;
    const float cend = cumh[CHUNK - 1];
    const int kb8 = (lane >> 4) * 8, kb16 = (lane >> 4) * 16, l15 = lane & 15;

    v8f zero = {};
    v8f acc[2][2];
    acc[0][0] = zero; acc[0][1] = zero; acc[1][0] = zero; acc[1][1] = zero;

    for (int kb = 0; kb < CHUNK; kb += 32) {
#pragma unroll
        for (int q = 0; q < 16; ++q) {          // weighted x^T slab
            const int idx = tid + q * 256;
            const int j = idx >> 7, p = idx & 127;
            const int l = kb + j;
            const float w = dt[(cb + l) * NHEADS + h] * __expf(cend - cumh[l]);
            aS[p][j] = (_Float16)((float)xf[(cb + l) * (long)D_INNER + h * HEADDIM + p] * w);
        }
#pragma unroll
        for (int q = 0; q < 8; ++q) {           // B^T slab [n][j]
            const int idx = tid + q * 256;
            const int j = idx >> 6, n = idx & 63;
            bS[n][j] = Bf[(cb + kb + j) * (long)D_STATE + n];
        }
        __syncthreads();
#pragma unroll
        for (int mt = 0; mt < 2; ++mt) {
            const int row = wm * 32 + mt * 16 + l15;
            v16h a;
#pragma unroll
            for (int i = 0; i < 16; ++i) a[i] = aS[row][kb16 + i];
#pragma unroll
            for (int nt = 0; nt < 2; ++nt) {
                const int col = wn * 32 + nt * 16 + l15;
                v16h b;
#pragma unroll
                for (int i = 0; i < 16; ++i) b[i] = bS[col][kb16 + i];
                acc[mt][nt] = __builtin_amdgcn_wmma_f32_16x16x32_f16(
                    false, a, false, b, (short)0, acc[mt][nt], false, false);
            }
        }
        __syncthreads();
    }
    float* out = states + ((long)bc * NHEADS + h) * (HEADDIM * D_STATE);
#pragma unroll
    for (int mt = 0; mt < 2; ++mt)
#pragma unroll
    for (int nt = 0; nt < 2; ++nt) {
        const int n     = wn * 32 + nt * 16 + l15;
        const int pbase = wm * 32 + mt * 16 + (lane >> 4) * 8;
#pragma unroll
        for (int r = 0; r < 8; ++r)
            out[(long)(pbase + r) * D_STATE + n] = acc[mt][nt][r];
    }
}

// ---------------------------------------------------------------------------
// 9) Sequential inter-chunk recurrence (per batch, head):
//    prev[c] = carry;  carry = carry * exp(cumEnd[c]) + states[c]
// ---------------------------------------------------------------------------
__global__ __launch_bounds__(256) void scan_kernel(
    const float* __restrict__ states, const float* __restrict__ cum,
    float* __restrict__ prev)
{
    const int b = blockIdx.x >> 4;
    const int h = blockIdx.x & 15;
    const int tid = threadIdx.x;
    float carry[32];
#pragma unroll
    for (int k = 0; k < 32; ++k) carry[k] = 0.0f;
    for (int c = 0; c < NCHUNK; ++c) {
        const int bc = b * NCHUNK + c;
        const long base = ((long)bc * NHEADS + h) * (HEADDIM * D_STATE);
        const float dec = __expf(cum[((long)bc * NHEADS + h) * CHUNK + CHUNK - 1]);
#pragma unroll
        for (int k = 0; k < 32; ++k) {
            const int e = tid + k * 256;
            prev[base + e] = carry[k];
            carry[k] = carry[k] * dec + states[base + e];
        }
    }
}

// ---------------------------------------------------------------------------
// 10) y_inter: y[l,p] += exp(cum[l]) * sum_n C[l,n] * prev[p,n]
//     M=128(l per block-half), N=128(p), K=64(n). Direct global fragments.
// ---------------------------------------------------------------------------
__global__ __launch_bounds__(256) void y_inter_kernel(
    const _Float16* __restrict__ Cf,   // [MTOT][D_STATE]
    const float* __restrict__ prev,    // [NBC][NHEADS][HEADDIM][D_STATE]
    const float* __restrict__ cum,
    float* __restrict__ y)
{
    const int mhalf = blockIdx.x, bc = blockIdx.y, h = blockIdx.z;
    const int tid = threadIdx.x, lane = tid & 31, wave = tid >> 5;
    const int wm = wave >> 1, wn = wave & 1;
    const int mbase = mhalf * 128;
    const long cb = (long)bc * CHUNK;
    const float* cumh = cum + ((long)bc * NHEADS + h) * CHUNK;
    const float* ph   = prev + ((long)bc * NHEADS + h) * (HEADDIM * D_STATE);
    const int kb8 = (lane >> 4) * 8, kb16 = (lane >> 4) * 16, l15 = lane & 15;

    v8f zero = {};
    v8f acc[2][4];
#pragma unroll
    for (int i = 0; i < 2; ++i)
#pragma unroll
    for (int j = 0; j < 4; ++j) acc[i][j] = zero;

#pragma unroll
    for (int kb = 0; kb < D_STATE; kb += 32) {
#pragma unroll
        for (int mt = 0; mt < 2; ++mt) {
            const int gi = mbase + wm * 32 + mt * 16 + l15;
            const _Float16* ar = Cf + (cb + gi) * (long)D_STATE + kb;
            v16h a;
#pragma unroll
            for (int i = 0; i < 16; ++i) a[i] = ar[kb16 + i];
#pragma unroll
            for (int nt = 0; nt < 4; ++nt) {
                const int gp = wn * 64 + nt * 16 + l15;
                v16h b;
#pragma unroll
                for (int i = 0; i < 16; ++i) b[i] = (_Float16)ph[(long)gp * D_STATE + kb + kb16 + i];
                acc[mt][nt] = __builtin_amdgcn_wmma_f32_16x16x32_f16(
                    false, a, false, b, (short)0, acc[mt][nt], false, false);
            }
        }
    }
#pragma unroll
    for (int mt = 0; mt < 2; ++mt)
#pragma unroll
    for (int nt = 0; nt < 4; ++nt) {
        const int gp    = wn * 64 + nt * 16 + l15;
        const int rbase = mbase + wm * 32 + mt * 16 + (lane >> 4) * 8;
#pragma unroll
        for (int r = 0; r < 8; ++r) {
            const int gi = rbase + r;
            const float es = __expf(cumh[gi]);
            y[(cb + gi) * (long)D_INNER + h * HEADDIM + gp] += es * acc[mt][nt][r];
        }
    }
}

// ---------------------------------------------------------------------------
// 11) g = (y + D*x) * silu(z);  g *= rsqrt(mean(g^2)+eps) * rms_w;  cast f16
// ---------------------------------------------------------------------------
__global__ __launch_bounds__(256) void gate_rms_kernel(
    const float* __restrict__ y, const _Float16* __restrict__ xf,
    const float* __restrict__ zx, const float* __restrict__ Dp,
    const float* __restrict__ rmsw, _Float16* __restrict__ g)
{
    __shared__ float red[256];
    const int row = blockIdx.x;
    const int tid = threadIdx.x;
    float vals[D_INNER / 256];
    float ss = 0.0f;
#pragma unroll
    for (int k = 0; k < D_INNER / 256; ++k) {
        const int c = tid + k * 256;
        const int h = c >> 7;
        const float xv = (float)xf[(long)row * D_INNER + c];
        const float yv = y[(long)row * D_INNER + c] + Dp[h] * xv;
        const float zv = zx[(long)row * NPAD_IN + c];
        const float gv = yv * siluf_(zv);
        vals[k] = gv;
        ss += gv * gv;
    }
    red[tid] = ss; __syncthreads();
    for (int o = 128; o > 0; o >>= 1) { if (tid < o) red[tid] += red[tid + o]; __syncthreads(); }
    const float rs = rsqrtf(red[0] * (1.0f / D_INNER) + RMS_EPS);
#pragma unroll
    for (int k = 0; k < D_INNER / 256; ++k) {
        const int c = tid + k * 256;
        g[(long)row * D_INNER + c] = (_Float16)(vals[k] * rs * rmsw[c]);
    }
}

// ---------------------------------------------------------------------------
// Launch
// ---------------------------------------------------------------------------
extern "C" void kernel_launch(void* const* d_in, const int* in_sizes, int n_in,
                              void* d_out, int out_size, void* d_ws, size_t ws_size,
                              hipStream_t stream)
{
    (void)in_sizes; (void)n_in; (void)out_size; (void)ws_size;
    const float* u        = (const float*)d_in[0];
    const float* in_proj  = (const float*)d_in[1];
    const float* conv_w   = (const float*)d_in[2];
    const float* conv_b   = (const float*)d_in[3];
    const float* dt_bias  = (const float*)d_in[4];
    const float* A_log    = (const float*)d_in[5];
    const float* Dp       = (const float*)d_in[6];
    const float* xnw      = (const float*)d_in[7];
    const float* xnb      = (const float*)d_in[8];
    const float* rmsw     = (const float*)d_in[9];
    const float* out_proj = (const float*)d_in[10];

    char* ws = (char*)d_ws;
    size_t off = 0;
    auto carve = [&](size_t bytes) { size_t o = off; off += (bytes + 255) & ~(size_t)255; return o; };

    _Float16* uln    = (_Float16*)(ws + carve((size_t)MTOT * DIM * 2));
    _Float16* win    = (_Float16*)(ws + carve((size_t)NPAD_IN * DIM * 2));
    float*    zx     = (float*)   (ws + carve((size_t)MTOT * NPAD_IN * 4));
    _Float16* xf     = (_Float16*)(ws + carve((size_t)MTOT * D_INNER * 2));
    _Float16* Bf     = (_Float16*)(ws + carve((size_t)MTOT * D_STATE * 2));
    _Float16* Cf     = (_Float16*)(ws + carve((size_t)MTOT * D_STATE * 2));
    float*    dtb    = (float*)   (ws + carve((size_t)MTOT * NHEADS * 4));
    float*    cum    = (float*)   (ws + carve((size_t)NBC * NHEADS * CHUNK * 4));
    float*    scores = (float*)   (ws + carve((size_t)NBC * CHUNK * CHUNK * 4));
    float*    states = (float*)   (ws + carve((size_t)NBC * NHEADS * HEADDIM * D_STATE * 4));
    float*    prev   = (float*)   (ws + carve((size_t)NBC * NHEADS * HEADDIM * D_STATE * 4));
    float*    ybuf   = (float*)   (ws + carve((size_t)MTOT * D_INNER * 4));
    _Float16* gbuf   = (_Float16*)(ws + carve((size_t)MTOT * D_INNER * 2));
    _Float16* wout   = (_Float16*)(ws + carve((size_t)DIM * D_INNER * 2));

    const float in_scale = 1.0f / 32.0f;   // DIM^-0.5 folded into the weights

    // weights
    l2norm_cast_kernel<<<NPAD_IN, 256, 0, stream>>>(in_proj, win, D_IN_PROJ, DIM, in_scale);
    l2norm_cast_kernel<<<DIM, 256, 0, stream>>>(out_proj, wout, DIM, D_INNER, 1.0f);
    // layernorm
    layernorm_cast_kernel<<<MTOT, 256, 0, stream>>>(u, xnw, xnb, uln);
    // in-proj GEMM: 8192 x 4352 x 1024
    wmma_gemm_kernel<<<dim3(MTOT / 128, NPAD_IN / 64, 1), 256, 0, stream>>>(
        uln, win, zx, DIM, NPAD_IN, 0, 0, 0);
    // conv + silu + split, dt
    {
        long n = (long)MTOT * CONV_DIM;
        conv_silu_kernel<<<(unsigned)((n + 255) / 256), 256, 0, stream>>>(zx, conv_w, conv_b, xf, Bf, Cf);
    }
    dt_softplus_kernel<<<(MTOT * NHEADS) / 256, 256, 0, stream>>>(zx, dt_bias, dtb);
    // cumulative dA
    cumsum_kernel<<<dim3(NBC, NHEADS), CHUNK, 0, stream>>>(dtb, A_log, cum);
    // per-chunk scores = Cc · Bc^T : batched 256x256x64
    wmma_gemm_kernel<<<dim3(CHUNK / 128, CHUNK / 64, NBC), 256, 0, stream>>>(
        Cf, Bf, scores, D_STATE, CHUNK,
        (long)CHUNK * D_STATE, (long)CHUNK * D_STATE, (long)CHUNK * CHUNK);
    // intra-chunk y
    y_intra_kernel<<<dim3(2, NBC, NHEADS), 256, 0, stream>>>(scores, cum, dtb, xf, ybuf);
    // chunk states, inter-chunk recurrence, inter-chunk y
    states_kernel<<<dim3(NBC, NHEADS), 256, 0, stream>>>(Bf, xf, cum, dtb, states);
    scan_kernel<<<BATCH * NHEADS, 256, 0, stream>>>(states, cum, prev);
    y_inter_kernel<<<dim3(2, NBC, NHEADS), 256, 0, stream>>>(Cf, prev, cum, ybuf);
    // gate + rmsnorm
    gate_rms_kernel<<<MTOT, 256, 0, stream>>>(ybuf, xf, zx, Dp, rmsw, gbuf);
    // out-proj GEMM: 8192 x 1024 x 2048
    wmma_gemm_kernel<<<dim3(MTOT / 128, DIM / 64, 1), 256, 0, stream>>>(
        gbuf, wout, (float*)d_out, D_INNER, DIM, 0, 0, 0);
}
